// ChannelAttentionDeformConv_26079041421814
// MI455X (gfx1250) — matrix operand order repaired and checked by reading, not compiled, not run on hardware
//
#include <hip/hip_runtime.h>
#include <hip/hip_bf16.h>
#include <stdint.h>

typedef _Float16 half_t;
typedef __attribute__((ext_vector_type(16))) _Float16 v16h;
typedef __attribute__((ext_vector_type(8)))  float    v8f;

#define CC    256
#define HH    64
#define WW    64
#define HWSZ  4096            // HH*WW
#define BB    4
#define KKT   9               // 3x3 taps
#define KTOT  2304            // CC*KKT  (GEMM K)
#define NPIX  16384           // BB*HWSZ (GEMM N)
#define GG    32
#define CPG   8               // CC/GG
#define EPSV  1e-5f

// ---------------------------------------------------------------------------
// Kernel 0: weight f32 -> f16 (row-major [256][2304], matches reference reshape)
// ---------------------------------------------------------------------------
__global__ void wcvt_kernel(const float* __restrict__ w, half_t* __restrict__ wh, int n) {
  int i = blockIdx.x * blockDim.x + threadIdx.x;
  if (i < n) wh[i] = (half_t)w[i];
}

// ---------------------------------------------------------------------------
// Kernel 1: deformable bilinear unfold + channel-attention gate.
// One block per pixel, one thread per channel. Writes colsT[n][c*9+k] in f16
// (K-major per pixel -> WMMA B fragments load as contiguous dword pairs).
// ---------------------------------------------------------------------------
__global__ void __launch_bounds__(256)
unfold_att_kernel(const float* __restrict__ x,
                  const float* __restrict__ off,
                  const float* __restrict__ att_w,
                  const float* __restrict__ att_b,
                  half_t* __restrict__ colsT) {
  __shared__ float offs[2 * KKT];
  __shared__ float red[256];

  const int n  = blockIdx.x;            // pixel id = b*HWSZ + hw
  const int b  = n >> 12;
  const int hw = n & (HWSZ - 1);
  const int h  = hw >> 6;
  const int w  = hw & (WW - 1);
  const int c  = threadIdx.x;

  // stage the 18 per-pixel offsets once (offset layout [B, 9, 2, H, W])
  if (c < 2 * KKT) offs[c] = off[((size_t)b * 2 * KKT + c) * HWSZ + hw];
  __syncthreads();

  const float* xb = x + ((size_t)b * CC + c) * HWSZ;

  float samp[KKT];
  float xmax = -3.4e38f;
#pragma unroll
  for (int k = 0; k < KKT; ++k) {
    const int   ky = k / 3 - 1, kx = k % 3 - 1;
    const float py = (float)(h + ky) + offs[2 * k + 0];
    const float px = (float)(w + kx) + offs[2 * k + 1];
    const float fy = floorf(py), fx = floorf(px);
    const int   iy0 = (int)fy, ix0 = (int)fx;
    const int   iy1 = iy0 + 1, ix1 = ix0 + 1;
    const float wy1 = py - fy, wx1 = px - fx;
    const float wy0 = 1.f - wy1, wx0 = 1.f - wx1;
    const bool y0v = (iy0 >= 0) & (iy0 < HH), y1v = (iy1 >= 0) & (iy1 < HH);
    const bool x0v = (ix0 >= 0) & (ix0 < WW), x1v = (ix1 >= 0) & (ix1 < WW);
    const float v00 = (y0v && x0v) ? xb[iy0 * WW + ix0] : 0.f;
    const float v01 = (y0v && x1v) ? xb[iy0 * WW + ix1] : 0.f;
    const float v10 = (y1v && x0v) ? xb[iy1 * WW + ix0] : 0.f;
    const float v11 = (y1v && x1v) ? xb[iy1 * WW + ix1] : 0.f;
    const float s = v00 * (wy0 * wx0) + v01 * (wy0 * wx1)
                  + v10 * (wy1 * wx0) + v11 * (wy1 * wx1);
    samp[k] = s;
    xmax = fmaxf(xmax, s);
  }

  // attention: sigmoid( sum_c att_w[c] * max_k samp ) — block reduction over C
  red[c] = att_w[c] * xmax;
  __syncthreads();
  for (int s = 128; s > 0; s >>= 1) {
    if (c < s) red[c] += red[c + s];
    __syncthreads();
  }
  const float att = 1.f / (1.f + __expf(-(red[0] + att_b[0])));

  half_t* dst = colsT + (size_t)n * KTOT + c * KKT;
#pragma unroll
  for (int k = 0; k < KKT; ++k) dst[k] = (half_t)(samp[k] * att);
}

// ---------------------------------------------------------------------------
// Kernel 2: WMMA GEMM  out[256, 16384] = Wh[256, 2304] @ colsT^T
// Block tile 64(M) x 256(N); 8 waves, each wave computes 32x64 via 2x4 WMMAs.
// A fragment: row-major weight, per-lane dword index j + 4*((j>=4)+hi).
// B fragment: K-major colsT row, per-lane dword index j + 4*((j>=4)+hi).
// ---------------------------------------------------------------------------
__global__ void __launch_bounds__(256)
gemm_wmma_kernel(const half_t* __restrict__ Wh,
                 const half_t* __restrict__ colsT,
                 float* __restrict__ out) {
  const int lane  = threadIdx.x & 31;
  const int wave  = threadIdx.x >> 5;
  const int waveM = wave & 1;           // 2 waves along M
  const int waveN = wave >> 1;          // 4 waves along N
  const int Mbase = blockIdx.y * 64 + waveM * 32;
  const int Nbase = blockIdx.x * 256 + waveN * 64;
  const int l15   = lane & 15;
  const int hi    = lane >> 4;          // 0 | 1

  v8f acc[2][4] = {};

  int dA[8], dB[8];
#pragma unroll
  for (int j = 0; j < 8; ++j) {
    dA[j] = j + 4 * ((j >= 4 ? 1 : 0) + hi);   // A-frag dword offset in 32-K slab
    dB[j] = j + 4 * ((j >= 4 ? 1 : 0) + hi);   // B-frag dword offset in 32-K slab
  }

  const uint32_t* aRow[2];
  const uint32_t* bRow[4];
#pragma unroll
  for (int mt = 0; mt < 2; ++mt)
    aRow[mt] = (const uint32_t*)(Wh + (size_t)(Mbase + mt * 16 + l15) * KTOT);
#pragma unroll
  for (int nt = 0; nt < 4; ++nt)
    bRow[nt] = (const uint32_t*)(colsT + (size_t)(Nbase + nt * 16 + l15) * KTOT);

  for (int kb = 0; kb < KTOT; kb += 32) {
    const int kd = kb >> 1;               // dword offset of this K slab

    union { v16h v; uint32_t u[8]; } a[2], bf[4];
#pragma unroll
    for (int mt = 0; mt < 2; ++mt)
#pragma unroll
      for (int j = 0; j < 8; ++j) a[mt].u[j] = aRow[mt][kd + dA[j]];
#pragma unroll
    for (int nt = 0; nt < 4; ++nt)
#pragma unroll
      for (int j = 0; j < 8; ++j) bf[nt].u[j] = bRow[nt][kd + dB[j]];

    // prefetch next K slab of the activation stream
    if (kb + 32 < KTOT) {
#pragma unroll
      for (int nt = 0; nt < 4; ++nt)
        __builtin_prefetch((const void*)(bRow[nt] + kd + 16), 0, 3);
    }

#pragma unroll
    for (int mt = 0; mt < 2; ++mt)
#pragma unroll
      for (int nt = 0; nt < 4; ++nt)
        acc[mt][nt] = __builtin_amdgcn_wmma_f32_16x16x32_f16(
            false, a[mt].v, false, bf[nt].v,
            (short)0, acc[mt][nt], false, false);
  }

  // scatter D fragments into [B, C, H, W] output
#pragma unroll
  for (int mt = 0; mt < 2; ++mt)
#pragma unroll
    for (int nt = 0; nt < 4; ++nt) {
      const int nn = Nbase + nt * 16 + l15;
      const int bb = nn >> 12;
      const int hw = nn & (HWSZ - 1);
      float* op = out + (size_t)bb * CC * HWSZ + hw;
#pragma unroll
      for (int r = 0; r < 8; ++r) {
        const int o = Mbase + mt * 16 + hi * 8 + r;
        op[(size_t)o * HWSZ] = acc[mt][nt][r];
      }
    }
}

// ---------------------------------------------------------------------------
// Kernel 3: GroupNorm(32) + ReLU, in place on d_out. One block per (b, g).
// ---------------------------------------------------------------------------
__global__ void __launch_bounds__(256)
gnorm_kernel(float* __restrict__ out,
             const float* __restrict__ gamma,
             const float* __restrict__ beta) {
  __shared__ float rs[256], rs2[256];
  const int bg = blockIdx.x;
  const int b  = bg >> 5;
  const int g  = bg & 31;
  float* base = out + (size_t)b * CC * HWSZ + (size_t)g * CPG * HWSZ;
  const int NEL = CPG * HWSZ;  // 32768

  float s = 0.f, s2 = 0.f;
  for (int i = threadIdx.x; i < NEL; i += 256) {
    const float v = base[i];
    s += v; s2 += v * v;
  }
  rs[threadIdx.x] = s; rs2[threadIdx.x] = s2;
  __syncthreads();
  for (int t = 128; t > 0; t >>= 1) {
    if (threadIdx.x < t) { rs[threadIdx.x] += rs[threadIdx.x + t];
                           rs2[threadIdx.x] += rs2[threadIdx.x + t]; }
    __syncthreads();
  }
  const float mean = rs[0] / (float)NEL;
  const float var  = rs2[0] / (float)NEL - mean * mean;
  const float inv  = rsqrtf(var + EPSV);

  for (int i = threadIdx.x; i < NEL; i += 256) {
    const int cch = g * CPG + (i >> 12);
    const float y = (base[i] - mean) * inv * gamma[cch] + beta[cch];
    base[i] = fmaxf(y, 0.f);
  }
}

// ---------------------------------------------------------------------------
extern "C" void kernel_launch(void* const* d_in, const int* in_sizes, int n_in,
                              void* d_out, int out_size, void* d_ws, size_t ws_size,
                              hipStream_t stream) {
  const float* x     = (const float*)d_in[0];
  const float* off   = (const float*)d_in[1];
  const float* wgt   = (const float*)d_in[2];
  const float* att_w = (const float*)d_in[3];
  const float* att_b = (const float*)d_in[4];
  const float* gamma = (const float*)d_in[5];
  const float* beta  = (const float*)d_in[6];
  float* out = (float*)d_out;

  // workspace: colsT f16 [NPIX][KTOT] (75.5 MB) + weight f16 (1.2 MB)
  half_t* colsT = (half_t*)d_ws;
  half_t* wh    = (half_t*)((char*)d_ws + (size_t)NPIX * KTOT * sizeof(half_t));

  wcvt_kernel<<<(CC * KTOT + 255) / 256, 256, 0, stream>>>(wgt, wh, CC * KTOT);
  unfold_att_kernel<<<NPIX, 256, 0, stream>>>(x, off, att_w, att_b, colsT);
  gemm_wmma_kernel<<<dim3(NPIX / 256, CC / 64), 256, 0, stream>>>(wh, colsT, out);
  gnorm_kernel<<<BB * GG, 256, 0, stream>>>(out, gamma, beta);
}
